// NeuroSymbolicExpert_6914897347291
// MI455X (gfx1250) — matrix operand order repaired and checked by reading, not compiled, not run on hardware
//
#include <hip/hip_runtime.h>

typedef __attribute__((ext_vector_type(16))) _Float16 v16h;
typedef __attribute__((ext_vector_type(4)))  _Float16 v4h;
typedef __attribute__((ext_vector_type(8)))  float    v8f;
typedef __attribute__((ext_vector_type(4)))  float    vf4;

#define B_ 4
#define S_ 128
#define D_ 512
#define H_ 8
#define F_ 2048
#define R_ 8
#define DH_ 64
#define THRESH_ 0.1f

__device__ __forceinline__ v4h cvt4(vf4 f) {
  v4h h;
  h.x = (_Float16)f.x; h.y = (_Float16)f.y;
  h.z = (_Float16)f.z; h.w = (_Float16)f.w;
  return h;
}

// ---------------------------------------------------------------------------
// LayerNorm over D=512, optional residual add. grid = B*S rows, block = 256.
// ---------------------------------------------------------------------------
__global__ __launch_bounds__(256)
void ln_kernel(const float* __restrict__ x, const float* __restrict__ res,
               const float* __restrict__ g, const float* __restrict__ b,
               float* __restrict__ out) {
  const int row = blockIdx.x;
  const int t = threadIdx.x;
  const long base = (long)row * D_;
  float v0 = x[base + t];
  float v1 = x[base + t + 256];
  if (res) { v0 += res[base + t]; v1 += res[base + t + 256]; }
  __shared__ float ssum[256], ssq[256];
  ssum[t] = v0 + v1;
  ssq[t]  = v0 * v0 + v1 * v1;
  __syncthreads();
  for (int s = 128; s > 0; s >>= 1) {
    if (t < s) { ssum[t] += ssum[t + s]; ssq[t] += ssq[t + s]; }
    __syncthreads();
  }
  const float mean = ssum[0] * (1.0f / D_);
  const float var  = ssq[0] * (1.0f / D_) - mean * mean;
  const float rstd = rsqrtf(var + 1e-5f);
  out[base + t]       = (v0 - mean) * rstd * g[t]       + b[t];
  out[base + t + 256] = (v1 - mean) * rstd * g[t + 256] + b[t + 256];
}

// ---------------------------------------------------------------------------
// Generic batched WMMA GEMM:  C = act(scale * (A @ B) + bias)
//   A: [M,K] fp32, row stride lda.
//   B: if !TRANSB, element (k,n) = Bsrc[k*ldb + n]; if TRANSB, Bsrc[n*ldb + k].
//   Per blockIdx.z operand base offset: (z/zmod)*outer + (z%zmod)*inner.
// Block: 256 threads = 8 waves. Block tile 128(M) x 64(N), K-chunks of 64.
// Software-pipelined + double-buffered LDS: one barrier per chunk, next
// chunk's global loads in flight during WMMA. Each wave: 16x64 strip, one A
// fragment reused over 4 B fragments, 8 v_wmma per chunk.
// Requires M%128==0, N%64==0, K%64==0 (true at all call sites).
// Fragment layouts follow CDNA5 ISA 7.12.2 (wave32).
// ---------------------------------------------------------------------------
template <int TRANSB>
__global__ __launch_bounds__(256)
void gemm_wmma(const float* __restrict__ A, const float* __restrict__ Bsrc,
               const float* __restrict__ bias, float* __restrict__ C,
               int K, int lda, int ldb, int ldc,
               long aOuter, long aInner, long bOuter, long bInner,
               long cOuter, long cInner, int zmod,
               float scale, int relu) {
  const int z = blockIdx.z;
  A    += (long)(z / zmod) * aOuter + (long)(z % zmod) * aInner;
  Bsrc += (long)(z / zmod) * bOuter + (long)(z % zmod) * bInner;
  C    += (long)(z / zmod) * cOuter + (long)(z % zmod) * cInner;

  __shared__ __align__(16) _Float16 As[2][128][64];  // [buf][m][k]
  __shared__ __align__(16) _Float16 Bt[2][64][64];   // [buf][n][k] (transposed)

  const int tid  = threadIdx.x;
  const int wave = tid >> 5;                // 0..7 -> 16-row strip
  const int lane = tid & 31;
  const int half = lane >> 4;               // lanes 0-15 vs 16-31
  const int l15  = lane & 15;
  const int m0 = blockIdx.x * 128;
  const int n0 = blockIdx.y * 64;

  // unified staging indices: r_ = row-in-tile base, c_ = 4-element col group
  const int r_ = tid >> 4;                  // 0..15 (stride 16 per j)
  const int c_ = (tid & 15) * 4;            // 0,4,..,60

  const v8f vz = {0.f, 0.f, 0.f, 0.f, 0.f, 0.f, 0.f, 0.f};
  v8f acc[4] = {vz, vz, vz, vz};
  vf4 ra[8], rb[4];

  // ---- prologue: load chunk 0 into registers ----
#pragma unroll
  for (int j = 0; j < 8; ++j)
    ra[j] = *(const vf4*)&A[(long)(m0 + r_ + j * 16) * lda + c_];
#pragma unroll
  for (int j = 0; j < 4; ++j)
    rb[j] = TRANSB ? *(const vf4*)&Bsrc[(long)(n0 + r_ + j * 16) * ldb + c_]
                   : *(const vf4*)&Bsrc[(long)(r_ + j * 16) * ldb + n0 + c_];

  int buf = 0;
  for (int kk = 0; kk < K; kk += 64) {
    // ---- stage registers -> LDS (buf) ----
#pragma unroll
    for (int j = 0; j < 8; ++j)
      *(v4h*)&As[buf][r_ + j * 16][c_] = cvt4(ra[j]);
    if (TRANSB) {
#pragma unroll
      for (int j = 0; j < 4; ++j)
        *(v4h*)&Bt[buf][r_ + j * 16][c_] = cvt4(rb[j]);
    } else {
#pragma unroll
      for (int j = 0; j < 4; ++j) {
        const int k = r_ + j * 16;
        Bt[buf][c_ + 0][k] = (_Float16)rb[j].x;
        Bt[buf][c_ + 1][k] = (_Float16)rb[j].y;
        Bt[buf][c_ + 2][k] = (_Float16)rb[j].z;
        Bt[buf][c_ + 3][k] = (_Float16)rb[j].w;
      }
    }
    __syncthreads();

    // ---- issue next chunk's global loads (in flight during WMMA) ----
    const int kn = kk + 64;
    if (kn < K) {
#pragma unroll
      for (int j = 0; j < 8; ++j)
        ra[j] = *(const vf4*)&A[(long)(m0 + r_ + j * 16) * lda + kn + c_];
#pragma unroll
      for (int j = 0; j < 4; ++j)
        rb[j] = TRANSB ? *(const vf4*)&Bsrc[(long)(n0 + r_ + j * 16) * ldb + kn + c_]
                       : *(const vf4*)&Bsrc[(long)(kn + r_ + j * 16) * ldb + n0 + c_];
      if (kn + 64 < K) {  // distant prefetch of chunk k+2
        __builtin_prefetch(&A[(long)(m0 + r_) * lda + kn + 64 + c_], 0, 1);
        __builtin_prefetch(TRANSB ? &Bsrc[(long)(n0 + r_) * ldb + kn + 64 + c_]
                                  : &Bsrc[(long)(kn + 64 + r_) * ldb + n0 + c_], 0, 1);
      }
    }

    // ---- compute: 2 K-subchunks x 4 N-tiles = 8 wmma ----
    union U { v16h v; vf4 f4[2]; };
    const _Float16* arow = &As[buf][wave * 16 + l15][0];
    const _Float16* brow0 = &Bt[buf][l15][0];
#pragma unroll
    for (int s = 0; s < 2; ++s) {
      // A fragment: e=0..7 -> K=8*half+e; e=8..15 -> K=16+8*half+(e-8)
      U af;
      af.f4[0] = *(const vf4*)(arow + s * 32 + 8 * half);
      af.f4[1] = *(const vf4*)(arow + s * 32 + 16 + 8 * half);
#pragma unroll
      for (int t = 0; t < 4; ++t) {
        // B fragment: e=0..7 -> K=8*half+e; e=8..15 -> K=16+8*half+(e-8)
        const _Float16* brow = brow0 + (t * 16) * 64;
        U bf;
        bf.f4[0] = *(const vf4*)(brow + s * 32 + 8 * half);
        bf.f4[1] = *(const vf4*)(brow + s * 32 + 16 + 8 * half);
        acc[t] = __builtin_amdgcn_wmma_f32_16x16x32_f16(false, af.v, false, bf.v,
                                                        (short)0, acc[t], false, false);
      }
    }
    buf ^= 1;   // ping-pong; single barrier per chunk is safe (see analysis)
  }

  // C/D layout: VGPR r, lane -> (M = r + 8*half, N = l15)
#pragma unroll
  for (int t = 0; t < 4; ++t) {
    const int col = n0 + t * 16 + l15;
    const float bvv = bias ? bias[col] : 0.f;
#pragma unroll
    for (int r = 0; r < 8; ++r) {
      const int row = m0 + wave * 16 + r + 8 * half;
      float v = acc[t][r] * scale + bvv;
      if (relu) v = fmaxf(v, 0.f);
      C[(long)row * ldc + col] = v;
    }
  }
}

// ---------------------------------------------------------------------------
// Softmax over last dim (S=128). grid = B*H*S rows, block = 128. In place.
// ---------------------------------------------------------------------------
__global__ __launch_bounds__(128)
void softmax_kernel(float* __restrict__ attn) {
  const long base = (long)blockIdx.x * S_;
  const int t = threadIdx.x;
  const float v = attn[base + t];
  __shared__ float red[128];
  red[t] = v;
  __syncthreads();
  for (int s = 64; s > 0; s >>= 1) {
    if (t < s) red[t] = fmaxf(red[t], red[t + s]);
    __syncthreads();
  }
  const float mx = red[0];
  __syncthreads();
  const float e = __expf(v - mx);
  red[t] = e;
  __syncthreads();
  for (int s = 64; s > 0; s >>= 1) {
    if (t < s) red[t] += red[t + s];
    __syncthreads();
  }
  attn[base + t] = e / red[0];
}

// ---------------------------------------------------------------------------
// am[b,i,j] = mean over heads of attn[b,h,i,j]
// ---------------------------------------------------------------------------
__global__ __launch_bounds__(256)
void am_kernel(const float* __restrict__ attn, float* __restrict__ am) {
  const int idx = blockIdx.x * 256 + threadIdx.x;
  if (idx >= B_ * S_ * S_) return;
  const int b = idx / (S_ * S_);
  const int ij = idx - b * S_ * S_;
  float s = 0.f;
#pragma unroll
  for (int h = 0; h < H_; ++h)
    s += attn[(long)(b * H_ + h) * (S_ * S_) + ij];
  am[idx] = s * (1.0f / H_);
}

// ---------------------------------------------------------------------------
// Fused pair relation classifier: one wave per (b,u,v) pair.
//   h = relu(a_part[b,u,:] + b_part[b,v,:] + rc_b1)      (never materialized)
//   logits = h @ rc_w2 + rc_b2 ; pred = argmax
//   edge[b,u,v] = pred if (am>thresh && u!=v && pred!=0) else 0
// ---------------------------------------------------------------------------
__global__ __launch_bounds__(256)
void pair_kernel(const float* __restrict__ a_part, const float* __restrict__ b_part,
                 const float* __restrict__ rc_b1, const float* __restrict__ rc_w2,
                 const float* __restrict__ rc_b2, const float* __restrict__ am,
                 int* __restrict__ edge) {
  const int wave = threadIdx.x >> 5;
  const int lane = threadIdx.x & 31;
  const int pair = blockIdx.x * 8 + wave;   // < B*S*S
  const int b  = pair >> 14;
  const int uv = pair & 16383;
  const int u = uv >> 7;
  const int v = uv & 127;
  const float* ap = a_part + (long)(b * S_ + u) * D_;
  const float* bp = b_part + (long)(b * S_ + v) * D_;
  float accr[R_] = {0, 0, 0, 0, 0, 0, 0, 0};
  for (int e = lane; e < D_; e += 32) {
    float hv = ap[e] + bp[e] + rc_b1[e];
    hv = fmaxf(hv, 0.f);
    const float* wr = rc_w2 + (long)e * R_;   // contiguous 8 floats
#pragma unroll
    for (int r = 0; r < R_; ++r) accr[r] += hv * wr[r];
  }
#pragma unroll
  for (int off = 16; off > 0; off >>= 1) {
#pragma unroll
    for (int r = 0; r < R_; ++r) accr[r] += __shfl_xor(accr[r], off, 32);
  }
  if (lane == 0) {
    int best = 0;
    float bestv = accr[0] + rc_b2[0];
#pragma unroll
    for (int r = 1; r < R_; ++r) {
      const float lv = accr[r] + rc_b2[r];
      if (lv > bestv) { bestv = lv; best = r; }   // strict > keeps first max
    }
    const bool adj = (am[(long)b * S_ * S_ + uv] > THRESH_) && (u != v);
    edge[(long)b * S_ * S_ + uv] = (adj && best != 0) ? best : 0;
  }
}

// ---------------------------------------------------------------------------
// Sparse RGCN aggregation + residual:
//   reasoned[b,v,:] = nu[b,v,:] + sum_{u: edge[b,u,v]=r>0} transformed[b,u,r,:]
// grid = B*S (one block per destination node v), block = 256.
// ---------------------------------------------------------------------------
__global__ __launch_bounds__(256)
void agg_kernel(const float* __restrict__ nu, const float* __restrict__ transformed,
                const int* __restrict__ edge, float* __restrict__ reasoned) {
  const int bv = blockIdx.x;      // b*S + v
  const int b = bv >> 7;
  const int v = bv & 127;
  __shared__ int er[S_];
  if (threadIdx.x < S_)
    er[threadIdx.x] = edge[((long)b * S_ + threadIdx.x) * S_ + v];
  __syncthreads();
  for (int e = threadIdx.x; e < D_; e += 256) {
    float acc = nu[(long)bv * D_ + e];
    for (int u = 0; u < S_; ++u) {
      const int r = er[u];
      if (r) acc += transformed[(((long)(b * S_ + u)) * R_ + r) * D_ + e];
    }
    reasoned[(long)bv * D_ + e] = acc;
  }
}

// ---------------------------------------------------------------------------
extern "C" void kernel_launch(void* const* d_in, const int* in_sizes, int n_in,
                              void* d_out, int out_size, void* d_ws, size_t ws_size,
                              hipStream_t stream) {
  (void)in_sizes; (void)n_in; (void)out_size; (void)ws_size;
  const float* x     = (const float*)d_in[0];
  const float* ln1_g = (const float*)d_in[1];
  const float* ln1_b = (const float*)d_in[2];
  const float* wq = (const float*)d_in[3];
  const float* bq = (const float*)d_in[4];
  const float* wk = (const float*)d_in[5];
  const float* bk = (const float*)d_in[6];
  const float* wv = (const float*)d_in[7];
  const float* bv = (const float*)d_in[8];
  const float* wo = (const float*)d_in[9];
  const float* bo = (const float*)d_in[10];
  const float* w1 = (const float*)d_in[11];
  const float* b1 = (const float*)d_in[12];
  const float* w2 = (const float*)d_in[13];
  const float* b2 = (const float*)d_in[14];
  const float* rc_w1 = (const float*)d_in[15];
  const float* rc_b1 = (const float*)d_in[16];
  const float* rc_w2 = (const float*)d_in[17];
  const float* rc_b2 = (const float*)d_in[18];
  const float* kg_w  = (const float*)d_in[19];
  const float* s2n_w = (const float*)d_in[20];
  const float* s2n_b = (const float*)d_in[21];
  const float* ln2_g = (const float*)d_in[22];
  const float* ln2_b = (const float*)d_in[23];
  float* out = (float*)d_out;

  // workspace layout (fp32 elements); peak ~26.7 MB
  float* W = (float*)d_ws;
  float* xn       = W + 0;        // 262144
  float* q        = W + 262144;   // 262144  [B,S,H,DH]
  float* kbuf     = W + 524288;   // 262144
  float* vbuf     = W + 786432;   // 262144
  float* attn     = W + 1048576;  // 524288  [B,H,S,S]
  float* am       = W + 1572864;  // 65536   [B,S,S]
  float* ctx      = W + 1638400;  // 262144
  float* attn_out = W + 1900544;  // 262144
  float* ffn1     = W + 2162688;  // 1048576 [B,S,F]
  float* nu       = W + 3211264;  // 262144
  float* a_part   = W + 3473408;  // 262144
  float* b_part   = W + 3735552;  // 262144
  float* transf   = W + 3997696;  // 2097152 [B,S,R,D]
  float* reasoned = W + 6094848;  // 262144
  float* neural   = W + 6356992;  // 262144
  int*   edge     = (int*)(W + 6619136); // 65536 ints

  const dim3 blk(256);

  // 1. xn = LN1(x)
  ln_kernel<<<B_ * S_, 256, 0, stream>>>(x, nullptr, ln1_g, ln1_b, xn);

  // 2. Q/K/V projections: [512,512] @ [512,512] + bias
  const dim3 g512(512 / 128, 512 / 64, 1);
  gemm_wmma<0><<<g512, blk, 0, stream>>>(xn, wq, bq, q,    512, 512, 512, 512,
                                         0, 0, 0, 0, 0, 0, 1, 1.0f, 0);
  gemm_wmma<0><<<g512, blk, 0, stream>>>(xn, wk, bk, kbuf, 512, 512, 512, 512,
                                         0, 0, 0, 0, 0, 0, 1, 1.0f, 0);
  gemm_wmma<0><<<g512, blk, 0, stream>>>(xn, wv, bv, vbuf, 512, 512, 512, 512,
                                         0, 0, 0, 0, 0, 0, 1, 1.0f, 0);

  // 3. scores[b,h] = Q Kᵀ / sqrt(DH): z = b*H+h, TRANSB (K rows are N dim)
  const dim3 gsc(128 / 128, 128 / 64, B_ * H_);
  gemm_wmma<1><<<gsc, blk, 0, stream>>>(q, kbuf, nullptr, attn, DH_, 512, 512, S_,
                                        (long)S_ * 512, DH_,               // A: q[b,h]
                                        (long)S_ * 512, DH_,               // B: k[b,h]
                                        (long)H_ * S_ * S_, (long)S_ * S_, // C
                                        H_, 0.125f, 0);

  // 4. softmax rows + head-mean
  softmax_kernel<<<B_ * H_ * S_, 128, 0, stream>>>(attn);
  am_kernel<<<(B_ * S_ * S_ + 255) / 256, 256, 0, stream>>>(attn, am);

  // 5. ctx[b,h] = attn[b,h] @ V[b,h]: M=128, K=128, N=64
  const dim3 gcx(128 / 128, 64 / 64, B_ * H_);
  gemm_wmma<0><<<gcx, blk, 0, stream>>>(attn, vbuf, nullptr, ctx, S_, S_, 512, 512,
                                        (long)H_ * S_ * S_, (long)S_ * S_,
                                        (long)S_ * 512, DH_,
                                        (long)S_ * 512, DH_,
                                        H_, 1.0f, 0);

  // 6. attn_out = ctx @ wo + bo
  gemm_wmma<0><<<g512, blk, 0, stream>>>(ctx, wo, bo, attn_out, 512, 512, 512, 512,
                                         0, 0, 0, 0, 0, 0, 1, 1.0f, 0);

  // 7. FFN: relu(attn_out @ w1 + b1) @ w2 + b2 -> nu
  const dim3 gff1(512 / 128, F_ / 64, 1);
  gemm_wmma<0><<<gff1, blk, 0, stream>>>(attn_out, w1, b1, ffn1, 512, 512, F_, F_,
                                         0, 0, 0, 0, 0, 0, 1, 1.0f, 1);
  gemm_wmma<0><<<g512, blk, 0, stream>>>(ffn1, w2, b2, nu, F_, F_, 512, 512,
                                         0, 0, 0, 0, 0, 0, 1, 1.0f, 0);

  // 8. pair classifier halves: a_part = nu @ rc_w1[:D], b_part = nu @ rc_w1[D:]
  gemm_wmma<0><<<g512, blk, 0, stream>>>(nu, rc_w1, nullptr, a_part,
                                         512, 512, 512, 512,
                                         0, 0, 0, 0, 0, 0, 1, 1.0f, 0);
  gemm_wmma<0><<<g512, blk, 0, stream>>>(nu, rc_w1 + (long)D_ * D_, nullptr, b_part,
                                         512, 512, 512, 512,
                                         0, 0, 0, 0, 0, 0, 1, 1.0f, 0);

  // 9. transformed[b,u,r,:] = nu[b,u,:] @ kg_w[r]: z = r, C interleaved (ldc = R*D)
  const dim3 gtr(512 / 128, 512 / 64, R_);
  gemm_wmma<0><<<gtr, blk, 0, stream>>>(nu, kg_w, nullptr, transf,
                                        512, 512, 512, R_ * D_,
                                        0, 0, (long)D_ * D_, 0, (long)D_, 0,
                                        1, 1.0f, 0);

  // 10. fused pair logits + argmax + adjacency mask -> edge map
  pair_kernel<<<(B_ * S_ * S_) / 8, 256, 0, stream>>>(a_part, b_part, rc_b1,
                                                      rc_w2, rc_b2, am, edge);

  // 11. sparse RGCN aggregation + residual -> reasoned
  agg_kernel<<<B_ * S_, 256, 0, stream>>>(nu, transf, edge, reasoned);

  // 12. neural_result = reasoned @ s2n_w + s2n_b
  gemm_wmma<0><<<g512, blk, 0, stream>>>(reasoned, s2n_w, s2n_b, neural,
                                         512, 512, 512, 512,
                                         0, 0, 0, 0, 0, 0, 1, 1.0f, 0);

  // 13. out = LN2(x + neural_result)
  ln_kernel<<<B_ * S_, 256, 0, stream>>>(x, neural, ln2_g, ln2_b, out);
}
